// Block_6511170421614
// MI455X (gfx1250) — matrix operand order repaired and checked by reading, not compiled, not run on hardware
//
#include <hip/hip_runtime.h>

typedef __bf16 bf16;
typedef __attribute__((ext_vector_type(16))) __bf16 v16bf;
typedef __attribute__((ext_vector_type(8)))  float  v8f;

#define WMMA_BF16(a, b, c) \
    __builtin_amdgcn_wmma_f32_16x16x32_bf16(false, (a), false, (b), (short)0, (c), false, false)

// ---- async Global->LDS (CDNA5) with sync fallback ----
#if defined(__has_builtin)
#  if __has_builtin(__builtin_amdgcn_global_load_async_to_lds_b128)
#    define HAVE_ASYNC_LDS 1
#  endif
#endif
#ifndef HAVE_ASYNC_LDS
#  define HAVE_ASYNC_LDS 0
#endif

#if HAVE_ASYNC_LDS
#  if __has_builtin(__builtin_amdgcn_s_wait_asynccnt)
#    define WAIT_ASYNC(n) __builtin_amdgcn_s_wait_asynccnt(n)
#  else
#    define WAIT_ASYNC(n) asm volatile("s_wait_asynccnt %0" ::"i"(n) : "memory")
#  endif
#else
#  define WAIT_ASYNC(n)
#endif

// builtin prototype (from compiler diagnostic): (v4i AS1*, v4i AS3*, imm, imm)
typedef int vsi4 __attribute__((__vector_size__(4 * sizeof(int))));
typedef __attribute__((address_space(1))) vsi4* gptr4;
typedef __attribute__((address_space(3))) vsi4* lptr4;

__device__ __forceinline__ void g2l_b128(void* lds, const void* g) {
#if HAVE_ASYNC_LDS
    __builtin_amdgcn_global_load_async_to_lds_b128((gptr4)g, (lptr4)lds, 0, 0);
#else
    *(uint4*)lds = *(const uint4*)g;
#endif
}

union FragU { v16bf v; uint4 q[2]; };

__device__ __forceinline__ v8f zero8() {
    v8f z;
#pragma unroll
    for (int i = 0; i < 8; ++i) z[i] = 0.0f;
    return z;
}

// A fragment: 16x32 bf16 tile (M x K), row-major source with leading dim `ld`.
// ISA layout: lane l<16 -> row l, K {0..7,16..23}; lane l>=16 -> row l-16, K {8..15,24..31}.
__device__ __forceinline__ v16bf load_fragA(const bf16* p, int ld) {
    const int l = threadIdx.x & 31;
    const bf16* base = p + (size_t)(l & 15) * ld + ((l >> 4) << 3);
    FragU f;
    f.q[0] = *(const uint4*)(base);
    f.q[1] = *(const uint4*)(base + 16);
    return f.v;
}

// B fragment: 32x16 bf16 tile with B(k,n) = W[n][k], W row-major [N,K], leading dim `ld`.
// Dense-B layout: lane l<16 -> col l, K {0..7,16..23}; lane l>=16 -> col l-16, K {8..15,24..31}.
__device__ __forceinline__ v16bf load_fragB(const bf16* p, int ld) {
    const int l = threadIdx.x & 31;
    const bf16* base = p + (size_t)(l & 15) * ld + ((l >> 4) << 3);
    FragU f;
    f.q[0] = *(const uint4*)(base);
    f.q[1] = *(const uint4*)(base + 16);
    return f.v;
}

__global__ __launch_bounds__(256) void cvt_bf16_k(const float* __restrict__ in,
                                                  bf16* __restrict__ out, int n) {
    int i = blockIdx.x * 256 + threadIdx.x;
    if (i < n) out[i] = (bf16)in[i];
}

// ---------------------------------------------------------------------------
// WMMA GEMM with async double-buffered LDS staging:
//   D[M,N] = X[M,K] (bf16) * W[N,K]^T (bf16)
// Block: 256 thr = 8 waves; block tile 64x128; wave tile 32x32.
// Per k-step (k=32): stage X 64x32 (4KB) + W 128x32 (8KB) into LDS via
// global_load_async_to_lds_b128 (3 async ops/wave), compute from other buffer.
// EPI 0: scatter qkv -> q[B,H,N,D], k[B,H,N,D], vT[B,H,D,N]   (all bf16)
// EPI 1: out_bf16 = D + bias
// EPI 2: out_f32  = D + bias + resid
// ---------------------------------------------------------------------------
template<int EPI>
__global__ __launch_bounds__(256) void gemm_wmma_k(
    const bf16* __restrict__ X, const bf16* __restrict__ W,
    const float* __restrict__ bias, const float* __restrict__ resid,
    float* __restrict__ outF, bf16* __restrict__ outB,
    bf16* __restrict__ qb, bf16* __restrict__ kb, bf16* __restrict__ vtb,
    int M, int N, int K)
{
    __shared__ __align__(16) bf16 xs[2][64 * 32];
    __shared__ __align__(16) bf16 wsh[2][128 * 32];

    const int t = threadIdx.x;
    const int wid = t >> 5;
    const int mb = blockIdx.x * 64;
    const int nb = blockIdx.y * 128;
    const int lm0 = (wid & 1) * 32;   // wave tile inside block
    const int ln0 = (wid >> 1) * 32;

    // staging coords: 16B chunk per thread per instruction
    const int xr = t >> 2, xc = (t & 3) << 3;          // X: 64 rows x 32 cols
    auto stage = [&](int buf, int kk) {
        g2l_b128(&xs[buf][xr * 32 + xc], X + (size_t)(mb + xr) * K + kk + xc);
#pragma unroll
        for (int i = 0; i < 2; ++i) {                  // W: 128 rows x 32 cols
            int c = t + i * 256;
            int wr = c >> 2, wc = (c & 3) << 3;
            g2l_b128(&wsh[buf][wr * 32 + wc], W + (size_t)(nb + wr) * K + kk + wc);
        }
    };

    v8f acc[2][2];
#pragma unroll
    for (int i = 0; i < 2; ++i)
#pragma unroll
        for (int j = 0; j < 2; ++j) acc[i][j] = zero8();

    stage(0, 0);
    int buf = 0;
    for (int kk = 0; kk < K; kk += 32, buf ^= 1) {
        if (kk + 32 < K) {
            stage(buf ^ 1, kk + 32);   // prefetch next tile into other buffer
            WAIT_ASYNC(3);             // wait for current tile (3 older ops)
        } else {
            WAIT_ASYNC(0);
        }
        __syncthreads();               // all waves' slices visible
        v16bf a0 = load_fragA(&xs[buf][lm0 * 32], 32);
        v16bf a1 = load_fragA(&xs[buf][(lm0 + 16) * 32], 32);
        v16bf b0 = load_fragB(&wsh[buf][ln0 * 32], 32);
        v16bf b1 = load_fragB(&wsh[buf][(ln0 + 16) * 32], 32);
        acc[0][0] = WMMA_BF16(a0, b0, acc[0][0]);
        acc[0][1] = WMMA_BF16(a0, b1, acc[0][1]);
        acc[1][0] = WMMA_BF16(a1, b0, acc[1][0]);
        acc[1][1] = WMMA_BF16(a1, b1, acc[1][1]);
        __syncthreads();               // readers done before buffer is re-staged
    }

    const int l = t & 31;
    const int hl = l >> 4, cc = l & 15;   // C/D layout: row = r + 8*hl, col = cc
#pragma unroll
    for (int mf = 0; mf < 2; ++mf)
#pragma unroll
    for (int nf = 0; nf < 2; ++nf)
#pragma unroll
    for (int r = 0; r < 8; ++r) {
        float v = acc[mf][nf][r];
        int gm = mb + lm0 + mf * 16 + hl * 8 + r;
        int gn = nb + ln0 + nf * 16 + cc;
        if (EPI == 0) {
            int b = gm >> 10, tok = gm & 1023;
            int part = gn / 768, rem = gn % 768;
            int hh = rem >> 6, d = rem & 63;
            size_t bh = (size_t)b * 12 + hh;
            bf16 o = (bf16)v;
            if (part == 0)      qb[(bh * 1024 + tok) * 64 + d] = o;
            else if (part == 1) kb[(bh * 1024 + tok) * 64 + d] = o;
            else                vtb[(bh * 64 + d) * 1024 + tok] = o;
        } else if (EPI == 1) {
            outB[(size_t)gm * N + gn] = (bf16)(v + bias[gn]);
        } else {
            outF[(size_t)gm * N + gn] = v + bias[gn] + resid[(size_t)gm * N + gn];
        }
    }
}

// ---------------------------------------------------------------------------
// Flash attention: grid (16 qblocks, B*NH), 128 thr = 4 waves, 16 q-rows/wave.
// q,k: [B,H,1024,64] bf16; vt: [B,H,64,1024] bf16; o: [B,1024,768] bf16.
// ---------------------------------------------------------------------------
__global__ __launch_bounds__(128) void attn_k(
    const bf16* __restrict__ q, const bf16* __restrict__ k,
    const bf16* __restrict__ vt, bf16* __restrict__ o)
{
    __shared__ __align__(16) bf16 pls[4][16 * 64];   // per-wave P tile
    const int wid = threadIdx.x >> 5;
    const int l = threadIdx.x & 31;
    const int hl = l >> 4, cc = l & 15;
    const int bh = blockIdx.y;
    const int q0 = blockIdx.x * 64 + wid * 16;

    const bf16* Qp = q + ((size_t)bh * 1024 + q0) * 64;
    const bf16* Kp = k + (size_t)bh * 1024 * 64;
    const bf16* Vp = vt + (size_t)bh * 64 * 1024;

    v16bf qa0 = load_fragA(Qp, 64);        // Q features 0..31
    v16bf qa1 = load_fragA(Qp + 32, 64);   // Q features 32..63

    v8f O[4];
#pragma unroll
    for (int nf = 0; nf < 4; ++nf) O[nf] = zero8();
    float mrow[8], lrow[8];
#pragma unroll
    for (int r = 0; r < 8; ++r) { mrow[r] = -1e30f; lrow[r] = 0.0f; }

    for (int j = 0; j < 1024; j += 64) {
        if (j + 64 < 1024) {   // global_prefetch_b8 of next K/V tiles
            __builtin_prefetch(Kp + (size_t)(j + 64) * 64, 0, 0);
            __builtin_prefetch(Vp + (size_t)j + 64, 0, 0);
        }
        // S = Q * K^T over this 64-key tile
        v8f S[4];
#pragma unroll
        for (int nf = 0; nf < 4; ++nf) {
            v8f s = zero8();
            v16bf b0 = load_fragB(Kp + (size_t)(j + nf * 16) * 64, 64);
            v16bf b1 = load_fragB(Kp + (size_t)(j + nf * 16) * 64 + 32, 64);
            s = WMMA_BF16(qa0, b0, s);
            s = WMMA_BF16(qa1, b1, s);
            S[nf] = s;
        }
        // online softmax (rows live in 16-lane halves; xor-shuffle stays in half)
        float nm[8];
#pragma unroll
        for (int r = 0; r < 8; ++r) {
            float v = fmaxf(fmaxf(S[0][r], S[1][r]), fmaxf(S[2][r], S[3][r]));
#pragma unroll
            for (int d = 1; d < 16; d <<= 1) v = fmaxf(v, __shfl_xor(v, d, 32));
            nm[r] = fmaxf(mrow[r], v * 0.125f);
        }
#pragma unroll
        for (int nf = 0; nf < 4; ++nf)
#pragma unroll
            for (int r = 0; r < 8; ++r)
                S[nf][r] = __expf(S[nf][r] * 0.125f - nm[r]);
#pragma unroll
        for (int r = 0; r < 8; ++r) {
            float ssum = S[0][r] + S[1][r] + S[2][r] + S[3][r];
#pragma unroll
            for (int d = 1; d < 16; d <<= 1) ssum += __shfl_xor(ssum, d, 32);
            float alpha = __expf(mrow[r] - nm[r]);
            lrow[r] = lrow[r] * alpha + ssum;
            mrow[r] = nm[r];
            O[0][r] *= alpha; O[1][r] *= alpha; O[2][r] *= alpha; O[3][r] *= alpha;
        }
        // D-layout -> A-layout via LDS (uniform loop: barriers are safe)
        __syncthreads();
#pragma unroll
        for (int nf = 0; nf < 4; ++nf)
#pragma unroll
            for (int r = 0; r < 8; ++r)
                pls[wid][(hl * 8 + r) * 64 + nf * 16 + cc] = (bf16)S[nf][r];
        __syncthreads();
        v16bf pa0 = load_fragA(&pls[wid][0], 64);
        v16bf pa1 = load_fragA(&pls[wid][32], 64);
        // O += P * V  (B-frags from transposed V: contiguous rows of vt)
#pragma unroll
        for (int nf = 0; nf < 4; ++nf) {
            v16bf vb0 = load_fragB(Vp + (size_t)(nf * 16) * 1024 + j, 1024);
            v16bf vb1 = load_fragB(Vp + (size_t)(nf * 16) * 1024 + j + 32, 1024);
            O[nf] = WMMA_BF16(pa0, vb0, O[nf]);
            O[nf] = WMMA_BF16(pa1, vb1, O[nf]);
        }
    }
    const int b = bh / 12, hh = bh % 12;
#pragma unroll
    for (int nf = 0; nf < 4; ++nf)
#pragma unroll
        for (int r = 0; r < 8; ++r) {
            float v = O[nf][r] / lrow[r];
            int tok = q0 + hl * 8 + r;
            int gc = hh * 64 + nf * 16 + cc;
            o[((size_t)b * 1024 + tok) * 768 + gc] = (bf16)v;
        }
}

// ---------------------------------------------------------------------------
// LayerNorm (+ optional grouped rational P5/Q4) -> bf16. One block per row.
// ---------------------------------------------------------------------------
template<int CD, int PER, bool RAT, typename TIN>
__global__ __launch_bounds__(256) void ln_rat_k(
    const TIN* __restrict__ xin, const float* __restrict__ gam,
    const float* __restrict__ bet, const float* __restrict__ ra,
    const float* __restrict__ rb, bf16* __restrict__ out)
{
    __shared__ float swa[8], swb[8], sm2[2];
    __shared__ float sca[48], scb[32];
    const int row = blockIdx.x;
    const TIN* xr = xin + (size_t)row * CD;

    float xv[PER];
    float s = 0.0f, s2 = 0.0f;
#pragma unroll
    for (int i = 0; i < PER; ++i) {
        float v = (float)xr[threadIdx.x + i * 256];
        xv[i] = v; s += v; s2 += v * v;
    }
    if (RAT) {
        if (threadIdx.x < 48) sca[threadIdx.x] = ra[threadIdx.x];
        if (threadIdx.x < 32) scb[threadIdx.x] = rb[threadIdx.x];
    }
#pragma unroll
    for (int d = 16; d > 0; d >>= 1) {
        s  += __shfl_down(s, d, 32);
        s2 += __shfl_down(s2, d, 32);
    }
    int wid = threadIdx.x >> 5, lane = threadIdx.x & 31;
    if (lane == 0) { swa[wid] = s; swb[wid] = s2; }
    __syncthreads();
    if (threadIdx.x == 0) {
        float ta = 0, tb = 0;
        for (int i = 0; i < 8; ++i) { ta += swa[i]; tb += swb[i]; }
        float mean = ta / CD;
        float var = tb / CD - mean * mean;
        sm2[0] = mean;
        sm2[1] = rsqrtf(var + 1e-5f);
    }
    __syncthreads();
    float mean = sm2[0], rstd = sm2[1];
#pragma unroll
    for (int i = 0; i < PER; ++i) {
        int c = threadIdx.x + i * 256;
        float y = (xv[i] - mean) * rstd * gam[c] + bet[c];
        if (RAT) {
            int g = c / (CD / 8);
            const float* a  = &sca[g * 6];
            const float* bb = &scb[g * 4];
            float P = a[5];
            P = P * y + a[4]; P = P * y + a[3]; P = P * y + a[2];
            P = P * y + a[1]; P = P * y + a[0];
            float Qp = bb[3];
            Qp = Qp * y + bb[2]; Qp = Qp * y + bb[1]; Qp = Qp * y + bb[0];
            y = P / (1.0f + fabsf(Qp * y));
        }
        out[(size_t)row * CD + c] = (bf16)y;
    }
}

extern "C" void kernel_launch(void* const* d_in, const int* in_sizes, int n_in,
                              void* d_out, int out_size, void* d_ws, size_t ws_size,
                              hipStream_t stream)
{
    (void)in_sizes; (void)n_in; (void)out_size; (void)ws_size;
    const float* x      = (const float*)d_in[0];
    const float* ln1_g  = (const float*)d_in[1];
    const float* ln1_b  = (const float*)d_in[2];
    const float* w_qkv  = (const float*)d_in[3];
    const float* w_proj = (const float*)d_in[4];
    const float* b_proj = (const float*)d_in[5];
    const float* ln2_g  = (const float*)d_in[6];
    const float* ln2_b  = (const float*)d_in[7];
    const float* rat1_a = (const float*)d_in[8];
    const float* rat1_b = (const float*)d_in[9];
    const float* w_fc1  = (const float*)d_in[10];
    const float* b_fc1  = (const float*)d_in[11];
    const float* lnk_g  = (const float*)d_in[12];
    const float* lnk_b  = (const float*)d_in[13];
    const float* rat2_a = (const float*)d_in[14];
    const float* rat2_b = (const float*)d_in[15];
    const float* w_fc2  = (const float*)d_in[16];
    const float* b_fc2  = (const float*)d_in[17];

    const int M = 16384;  // B*N
    char* wsb = (char*)d_ws;
    size_t off = 0;
    auto take = [&](size_t bytes) -> void* {
        void* p = (void*)(wsb + off);
        off = (off + bytes + 255) & ~(size_t)255;
        return p;
    };
    bf16*  wq  = (bf16*)take((size_t)2304 * 768 * 2);
    bf16*  wp  = (bf16*)take((size_t)768 * 768 * 2);
    bf16*  wf1 = (bf16*)take((size_t)3072 * 768 * 2);
    bf16*  wf2 = (bf16*)take((size_t)768 * 3072 * 2);
    bf16*  hln = (bf16*)take((size_t)M * 768 * 2);   // LN1(x)
    bf16*  qb  = (bf16*)take((size_t)M * 768 * 2);   // [B,H,N,D]
    bf16*  kb  = (bf16*)take((size_t)M * 768 * 2);   // [B,H,N,D]
    bf16*  vtb = (bf16*)take((size_t)M * 768 * 2);   // [B,H,D,N]
    bf16*  ob  = (bf16*)take((size_t)M * 768 * 2);   // attention out [B,N,C]
    float* x1  = (float*)take((size_t)M * 768 * 4);  // x + proj (f32 residual)
    bf16*  ha1 = (bf16*)take((size_t)M * 768 * 2);   // rat1(LN2(x1))
    bf16*  h1  = (bf16*)take((size_t)M * 3072 * 2);  // fc1 out
    bf16*  h2  = (bf16*)take((size_t)M * 3072 * 2);  // rat2(LNk(h1))

    int nw;
    nw = 2304 * 768; cvt_bf16_k<<<(nw + 255) / 256, 256, 0, stream>>>(w_qkv, wq, nw);
    nw = 768 * 768;  cvt_bf16_k<<<(nw + 255) / 256, 256, 0, stream>>>(w_proj, wp, nw);
    nw = 3072 * 768; cvt_bf16_k<<<(nw + 255) / 256, 256, 0, stream>>>(w_fc1, wf1, nw);
    nw = 768 * 3072; cvt_bf16_k<<<(nw + 255) / 256, 256, 0, stream>>>(w_fc2, wf2, nw);

    // LN1
    ln_rat_k<768, 3, false, float><<<M, 256, 0, stream>>>(x, ln1_g, ln1_b, nullptr, nullptr, hln);
    // QKV GEMM with head-split / V-transpose scatter
    gemm_wmma_k<0><<<dim3(M / 64, 2304 / 128), 256, 0, stream>>>(
        hln, wq, nullptr, nullptr, nullptr, nullptr, qb, kb, vtb, M, 2304, 768);
    // Flash attention
    attn_k<<<dim3(1024 / 64, 16 * 12), 128, 0, stream>>>(qb, kb, vtb, ob);
    // proj + bias + residual -> x1 (f32)
    gemm_wmma_k<2><<<dim3(M / 64, 768 / 128), 256, 0, stream>>>(
        ob, wp, b_proj, x, x1, nullptr, nullptr, nullptr, nullptr, M, 768, 768);
    // LN2 + rational1
    ln_rat_k<768, 3, true, float><<<M, 256, 0, stream>>>(x1, ln2_g, ln2_b, rat1_a, rat1_b, ha1);
    // FC1 + bias -> bf16
    gemm_wmma_k<1><<<dim3(M / 64, 3072 / 128), 256, 0, stream>>>(
        ha1, wf1, b_fc1, nullptr, nullptr, h1, nullptr, nullptr, nullptr, M, 3072, 768);
    // LNk + rational2
    ln_rat_k<3072, 12, true, bf16><<<M, 256, 0, stream>>>(h1, lnk_g, lnk_b, rat2_a, rat2_b, h2);
    // FC2 + bias + residual(x1) -> d_out (f32)
    gemm_wmma_k<2><<<dim3(M / 64, 768 / 128), 256, 0, stream>>>(
        h2, wf2, b_fc2, x1, (float*)d_out, nullptr, nullptr, nullptr, nullptr, M, 768, 3072);
}
